// Mask2FormerHead_3186865734406
// MI455X (gfx1250) — matrix operand order repaired and checked by reading, not compiled, not run on hardware
//
#include <hip/hip_runtime.h>

// ---------------------------------------------------------------------------
// CDNA5 (gfx1250) Mask2Former head forward.
// All dense contractions run on V_WMMA_F32_16X16X32_BF16 (bf16 in, f32 acc),
// staged through LDS with vectorized/branchless edge handling, fused
// BN/bias/residual/ReLU epilogues.
// ---------------------------------------------------------------------------

typedef __attribute__((ext_vector_type(16))) __bf16 v16bf;
typedef __attribute__((ext_vector_type(8)))  float  v8f;

// ------------------------------- GEMM ---------------------------------------

struct GemmP {
  const __bf16* A;        // A[m*lda + k]           (weights or activations)
  const __bf16* B;        // normal: B[k*ldb+n]; btrans: B[n*ldb+k]; conv: act [IC][H][W]
  float*        C;        // optional f32 out
  __bf16*       Cb;       // optional bf16 out
  const float*  residual; // optional f32 residual  R[m*ldr + n]
  const float*  bias;     // per-M (mode 1) or per-N (mode 2)
  const float*  bnScale;  // per-M folded BN scale
  const float*  bnShift;  // per-M folded BN shift
  long long aZ, bZ, cZ, cbZ, rZ;  // blockIdx.z strides (elements)
  int lda, ldb, ldc, ldcb, ldr;
  int M, N, K;
  int btrans, biasMode, relu;
  int convH, convW, convKH, convKW, convPad; // convKH>0 -> implicit-GEMM conv
};

// Block tile: 64(M) x 128(N), BK=32. 8 waves in 2(M)x4(N); each wave owns a
// 32x32 C slab -> 4 accumulators, 4 WMMAs per K-chunk.
__global__ __launch_bounds__(256) void k_gemm(GemmP p) {
  __shared__ alignas(16) __bf16 As[64][32];   // As[mLocal][k]
  __shared__ alignas(16) __bf16 Bs[128][32];  // Bs[nLocal][k] (k-contiguous)
  const int tid  = threadIdx.x;
  const int lane = tid & 31;
  const int wave = tid >> 5;
  const int wm   = wave & 1;   // which 32-row half
  const int wn   = wave >> 1;  // which 32-col slab
  const int z    = blockIdx.z;
  const int m0   = blockIdx.y * 64;
  const int n0   = blockIdx.x * 128;

  const __bf16* A = p.A + (long long)z * p.aZ;
  const __bf16* B = p.B + (long long)z * p.bZ;

  const bool conv = (p.convKH > 0);
  const int taps = conv ? p.convKH * p.convKW : 1;
  const int IC   = p.K / taps;
  const int HW   = p.convH * p.convW;

  // staging roles
  const int aRow = tid >> 2;        // 0..63
  const int aKc  = (tid & 3) * 8;   // k chunk for A
  const int bKr  = tid >> 3;        // 0..31  (normal/conv B: k row)
  const int bNc  = (tid & 7) * 16;  // 16 contiguous n
  const int bNr  = tid >> 1;        // 0..127 (btrans B: n row)
  const int bKc  = (tid & 1) * 16;  // 16 contiguous k

  v8f c00 = {0.f,0.f,0.f,0.f,0.f,0.f,0.f,0.f};
  v8f c01 = {0.f,0.f,0.f,0.f,0.f,0.f,0.f,0.f};
  v8f c10 = {0.f,0.f,0.f,0.f,0.f,0.f,0.f,0.f};
  v8f c11 = {0.f,0.f,0.f,0.f,0.f,0.f,0.f,0.f};

  for (int tap = 0; tap < taps; ++tap) {
    int ky = 0, kx = 0;
    if (conv) { ky = tap / p.convKW; kx = tap % p.convKW; }
    for (int k0 = 0; k0 < IC; k0 += 32) {
      // ---------------- stage A tile (64 x 32) ----------------
      {
        const int  m   = m0 + aRow;
        const bool mok = (m < p.M);
        const int  mc  = mok ? m : (p.M - 1);
        if (conv) {
          // strided by taps; branchless clamped loads
#pragma unroll
          for (int j = 0; j < 8; ++j) {
            const int k = k0 + aKc + j;
            const __bf16 v = A[(long long)mc * p.lda + (long long)k * taps + tap];
            As[aRow][aKc + j] = mok ? v : (__bf16)0.f;
          }
        } else if (mok && (k0 + aKc + 8) <= p.K) {
          *(uint4*)(&As[aRow][aKc]) =
              *(const uint4*)(A + (long long)m * p.lda + k0 + aKc);
          if ((k0 + 32) < p.K)
            __builtin_prefetch(A + (long long)m * p.lda + k0 + 32 + aKc, 0, 1);
        } else {
#pragma unroll
          for (int j = 0; j < 8; ++j) {
            const int k   = k0 + aKc + j;
            const bool ok = mok && (k < p.K);
            const int kc2 = (k < p.K) ? k : (p.K - 1);
            const __bf16 v = A[(long long)mc * p.lda + kc2];
            As[aRow][aKc + j] = ok ? v : (__bf16)0.f;
          }
        }
      }
      // ---------------- stage B tile (32 x 128) ----------------
      if (conv) {
        // one channel row, 16 contiguous pixels; branchless pad handling
        const int k = k0 + bKr;
        const __bf16* brow = B + (long long)k * HW;
#pragma unroll
        for (int j = 0; j < 16; ++j) {
          const int n   = n0 + bNc + j;
          const bool nk = (n < p.N);
          const int nc2 = nk ? n : 0;
          const int y = nc2 / p.convW;
          const int x = nc2 - y * p.convW;
          const int ys = y + ky - p.convPad;
          const int xs = x + kx - p.convPad;
          const bool ok = nk && (ys >= 0) && (ys < p.convH) && (xs >= 0) && (xs < p.convW);
          const int ysc = min(max(ys, 0), p.convH - 1);
          const int xsc = min(max(xs, 0), p.convW - 1);
          const __bf16 v = brow[(long long)ysc * p.convW + xsc];
          Bs[bNc + j][bKr] = ok ? v : (__bf16)0.f;
        }
      } else if (p.btrans) {
        const int  n   = n0 + bNr;
        const bool nok = (n < p.N);
        if (nok && (k0 + bKc + 16) <= p.K) {
          const uint4* src = (const uint4*)(B + (long long)n * p.ldb + k0 + bKc);
          *(uint4*)(&Bs[bNr][bKc])     = src[0];
          *(uint4*)(&Bs[bNr][bKc + 8]) = src[1];
        } else {
          const int ncl = nok ? n : (p.N - 1);
#pragma unroll
          for (int j = 0; j < 16; ++j) {
            const int k   = k0 + bKc + j;
            const bool ok = nok && (k < p.K);
            const int kc2 = (k < p.K) ? k : (p.K - 1);
            const __bf16 v = B[(long long)ncl * p.ldb + kc2];
            Bs[bNr][bKc + j] = ok ? v : (__bf16)0.f;
          }
        }
      } else {
        // k-major source: load 16 contiguous n (coalesced b128), transpose in LDS
        const int  k    = k0 + bKr;
        const bool kok  = (k < p.K);
        const int  kcl  = kok ? k : (p.K - 1);
        if (kok && (n0 + bNc + 16) <= p.N) {
          const uint4* src = (const uint4*)(B + (long long)k * p.ldb + n0 + bNc);
          alignas(16) __bf16 tmp[16];
          *(uint4*)(&tmp[0]) = src[0];
          *(uint4*)(&tmp[8]) = src[1];
#pragma unroll
          for (int j = 0; j < 16; ++j) Bs[bNc + j][bKr] = tmp[j];
        } else {
#pragma unroll
          for (int j = 0; j < 16; ++j) {
            const int n   = n0 + bNc + j;
            const bool ok = kok && (n < p.N);
            const int ncl = (n < p.N) ? n : (p.N - 1);
            const __bf16 v = B[(long long)kcl * p.ldb + ncl];
            Bs[bNc + j][bKr] = ok ? v : (__bf16)0.f;
          }
        }
      }
      __syncthreads();

      // ---- fragments: contiguous k = 16*(lane/16) + s in both A and B ----
      const int mlc = lane & 15;
      const int kh  = lane >> 4;
      const __bf16* pa0 = &As[wm * 32 + mlc][0];
      const __bf16* pa1 = &As[wm * 32 + 16 + mlc][0];
      const __bf16* pb0 = &Bs[wn * 32 + mlc][kh * 16];
      const __bf16* pb1 = &Bs[wn * 32 + 16 + mlc][kh * 16];
      v16bf a0, a1, b0, b1;
#pragma unroll
      for (int i = 0; i < 8; ++i) {
        a0[i]     = pa0[kh * 16 + i];
        a0[i + 8] = pa0[kh * 16 + 8 + i];
        a1[i]     = pa1[kh * 16 + i];
        a1[i + 8] = pa1[kh * 16 + 8 + i];
      }
#pragma unroll
      for (int i = 0; i < 16; ++i) {
        b0[i] = pb0[i];
        b1[i] = pb1[i];
      }
      c00 = __builtin_amdgcn_wmma_f32_16x16x32_bf16(false, a0, false, b0, (short)0, c00, false, false);
      c01 = __builtin_amdgcn_wmma_f32_16x16x32_bf16(false, a0, false, b1, (short)0, c01, false, false);
      c10 = __builtin_amdgcn_wmma_f32_16x16x32_bf16(false, a1, false, b0, (short)0, c10, false, false);
      c11 = __builtin_amdgcn_wmma_f32_16x16x32_bf16(false, a1, false, b1, (short)0, c11, false, false);
      __syncthreads();
    }
  }

  // ---- fused epilogue ----
  const int mlc = lane & 15;
  const int kh  = lane >> 4;
  float*       Cz  = p.C        ? p.C        + (long long)z * p.cZ  : nullptr;
  __bf16*      Cbz = p.Cb       ? p.Cb       + (long long)z * p.cbZ : nullptr;
  const float* Rz  = p.residual ? p.residual + (long long)z * p.rZ  : nullptr;
  v8f accs[2][2] = {{c00, c01}, {c10, c11}};
#pragma unroll
  for (int t = 0; t < 2; ++t) {
    const int mb = m0 + wm * 32 + t * 16 + kh * 8;
#pragma unroll
    for (int u = 0; u < 2; ++u) {
      const int n = n0 + wn * 32 + u * 16 + mlc;
#pragma unroll
      for (int r = 0; r < 8; ++r) {
        const int m = mb + r;
        if (m < p.M && n < p.N) {
          float v = accs[t][u][r];
          if (p.bnScale) v = v * p.bnScale[m] + p.bnShift[m];
          if (p.biasMode == 1) v += p.bias[m];
          else if (p.biasMode == 2) v += p.bias[n];
          if (Rz) v += Rz[(long long)m * p.ldr + n];
          if (p.relu) v = fmaxf(v, 0.f);
          if (Cz)  Cz [(long long)m * p.ldc  + n] = v;
          if (Cbz) Cbz[(long long)m * p.ldcb + n] = (__bf16)v;
        }
      }
    }
  }
}

// --------------------------- support kernels --------------------------------

__global__ void k_cvt_bf16(const float* x, __bf16* y, long long n) {
  long long i = (long long)blockIdx.x * blockDim.x + threadIdx.x;
  if (i < n) y[i] = (__bf16)x[i];
}

__global__ void k_bn_prep(const float* g, const float* b, const float* m, const float* v,
                          float* scale, float* shift, int C) {
  int c = blockIdx.x * blockDim.x + threadIdx.x;
  if (c < C) {
    float s = g[c] * rsqrtf(v[c] + 1e-5f);
    scale[c] = s;
    shift[c] = b[c] - m[c] * s;
  }
}

__global__ void k_bcast_q(const float* qe, float* qf, __bf16* qb, int per, int total) {
  int i = blockIdx.x * blockDim.x + threadIdx.x;
  if (i < total) {
    float v = qe[i % per];
    qf[i] = v;
    qb[i] = (__bf16)v;
  }
}

// LayerNorm over D=256, token-major f32 in -> bf16 out
__global__ __launch_bounds__(256) void k_ln_rows(const float* x, const float* g,
                                                 const float* bb, __bf16* y) {
  __shared__ float red[256];
  const int t = threadIdx.x;
  const long long row = blockIdx.x;
  const float v = x[row * 256 + t];
  red[t] = v; __syncthreads();
  for (int o = 128; o > 0; o >>= 1) { if (t < o) red[t] += red[t + o]; __syncthreads(); }
  const float mean = red[0] * (1.f / 256.f);
  __syncthreads();
  const float d = v - mean;
  red[t] = d * d; __syncthreads();
  for (int o = 128; o > 0; o >>= 1) { if (t < o) red[t] += red[t + o]; __syncthreads(); }
  const float var = red[0] * (1.f / 256.f);
  y[row * 256 + t] = (__bf16)(d * rsqrtf(var + 1e-5f) * g[t] + bb[t]);
}

// LayerNorm over channels of a channel-major map f[C=256][P] -> token-major bf16
__global__ __launch_bounds__(256) void k_ln_cols(const float* f, long long zF,
                                                 const float* g, const float* bb,
                                                 __bf16* y, long long zY, int P) {
  __shared__ float red[256];
  const int c = threadIdx.x;
  const int p = blockIdx.x;
  const int z = blockIdx.y;
  const float v = f[(long long)z * zF + (long long)c * P + p];
  red[c] = v; __syncthreads();
  for (int o = 128; o > 0; o >>= 1) { if (c < o) red[c] += red[c + o]; __syncthreads(); }
  const float mean = red[0] * (1.f / 256.f);
  __syncthreads();
  const float d = v - mean;
  red[c] = d * d; __syncthreads();
  for (int o = 128; o > 0; o >>= 1) { if (c < o) red[c] += red[c + o]; __syncthreads(); }
  const float var = red[0] * (1.f / 256.f);
  y[(long long)z * zY + (long long)p * 256 + c] =
      (__bf16)(d * rsqrtf(var + 1e-5f) * g[c] + bb[c]);
}

// softmax over row of length L: v = s*scale + bias; writes bf16 probabilities
__global__ __launch_bounds__(256) void k_softmax(const float* S, __bf16* Aout,
                                                 long long sZ, int ld, int L, float scale,
                                                 const float* bias, long long bZ, int nh) {
  const int q = blockIdx.x, z = blockIdx.y, t = threadIdx.x;
  const float* row = S + (long long)z * sZ + (long long)q * ld;
  const float* br  = bias ? bias + (long long)(z / nh) * bZ + (long long)q * L : nullptr;
  __shared__ float red[256];
  float mx = -3.0e38f;
  for (int k = t; k < L; k += 256) {
    float v = row[k] * scale + (br ? br[k] : 0.f);
    mx = fmaxf(mx, v);
  }
  red[t] = mx; __syncthreads();
  for (int o = 128; o > 0; o >>= 1) { if (t < o) red[t] = fmaxf(red[t], red[t + o]); __syncthreads(); }
  mx = red[0]; __syncthreads();
  float sum = 0.f;
  for (int k = t; k < L; k += 256) {
    float v = row[k] * scale + (br ? br[k] : 0.f);
    sum += __expf(v - mx);
  }
  red[t] = sum; __syncthreads();
  for (int o = 128; o > 0; o >>= 1) { if (t < o) red[t] += red[t + o]; __syncthreads(); }
  const float inv = 1.f / red[0];
  __bf16* arow = Aout + (long long)z * sZ + (long long)q * ld;
  for (int k = t; k < L; k += 256)
    arow[k] = (__bf16)(__expf(row[k] * scale + (br ? br[k] : 0.f) - mx) * inv);
}

// out = a + nearest(src); writes f32 + bf16
__global__ void k_nearest_add(const float* a, const float* src, float* of, __bf16* ob,
                              int C, int h, int w, int sh, int sw) {
  long long i = (long long)blockIdx.x * blockDim.x + threadIdx.x;
  const long long total = (long long)C * h * w;
  if (i >= total) return;
  int c = (int)(i / (h * w));
  int p = (int)(i % (h * w));
  int y = p / w, x = p % w;
  int ys = (y * sh) / h, xs = (x * sw) / w;
  float v = a[i] + src[(long long)c * sh * sw + (long long)ys * sw + xs];
  of[i] = v;
  ob[i] = (__bf16)v;
}

// half-pixel bilinear; optional f32 and/or bf16 outputs
__global__ void k_bilinear(const float* x, float* of, __bf16* ob,
                           int C, int ih, int iw, int oh, int ow) {
  long long i = (long long)blockIdx.x * blockDim.x + threadIdx.x;
  const long long total = (long long)C * oh * ow;
  if (i >= total) return;
  int c = (int)(i / ((long long)oh * ow));
  int p = (int)(i % ((long long)oh * ow));
  int y = p / ow, xo = p % ow;
  float sy = (y + 0.5f) * (float)ih / (float)oh - 0.5f;
  float sx = (xo + 0.5f) * (float)iw / (float)ow - 0.5f;
  sy = fminf(fmaxf(sy, 0.f), (float)(ih - 1));
  sx = fminf(fmaxf(sx, 0.f), (float)(iw - 1));
  int y0 = (int)sy, x0 = (int)sx;
  int y1 = min(y0 + 1, ih - 1), x1 = min(x0 + 1, iw - 1);
  float fy = sy - y0, fx = sx - x0;
  const float* base = x + (long long)c * ih * iw;
  float v00 = base[(long long)y0 * iw + x0];
  float v01 = base[(long long)y0 * iw + x1];
  float v10 = base[(long long)y1 * iw + x0];
  float v11 = base[(long long)y1 * iw + x1];
  float v = v00 * (1 - fy) * (1 - fx) + v01 * (1 - fy) * fx +
            v10 * fy * (1 - fx) + v11 * fy * fx;
  if (of) of[i] = v;
  if (ob) ob[i] = (__bf16)v;
}

// attention bias from mask logits (128x128) nearest-sampled at (h,w)
__global__ __launch_bounds__(256) void k_mask_bias(const float* mlg, float* bias,
                                                   int h, int w) {
  const int q = blockIdx.x, b = blockIdx.y;
  const int row = b * 100 + q;
  const float* lrow = mlg + (long long)row * 16384;
  const int P = h * w;
  __shared__ int anyv;
  if (threadIdx.x == 0) anyv = 0;
  __syncthreads();
  int loc = 0;
  for (int p = threadIdx.x; p < P; p += 256) {
    int y = p / w, x = p % w;
    int yi = (y * 128) / h, xi = (x * 128) / w;
    if (lrow[yi * 128 + xi] > 0.f) loc = 1;
  }
  if (loc) atomicOr(&anyv, 1);
  __syncthreads();
  const int any = anyv;
  float* brow = bias + (long long)row * P;
  for (int p = threadIdx.x; p < P; p += 256) {
    int y = p / w, x = p % w;
    int yi = (y * 128) / h, xi = (x * 128) / w;
    bool m = lrow[yi * 128 + xi] > 0.f;
    brow[p] = (any && !m) ? -1e9f : 0.f;
  }
}

// ------------------------------ host side -----------------------------------

struct CbaPH { const float *w, *g, *b, *m, *v; };
struct ConvPH { const float *w, *b; };
struct LinPH { const float *w, *b; };
struct MhaPH { const float *wi, *bi, *wo, *bo; };
struct LnPH  { const float *g, *b; };
struct LayerPH { MhaPH xattn, sattn; LnPH nq, nkv, n1, n2; LinPH ff1, ff2; };

extern "C" void kernel_launch(void* const* d_in, const int* in_sizes, int n_in,
                              void* d_out, int out_size, void* d_ws, size_t ws_size,
                              hipStream_t stream) {
  (void)in_sizes; (void)n_in; (void)out_size; (void)ws_size;
  const int B = 2;

  // ---- input mapping: setup_inputs() insertion order ----
  int ii = 0;
  auto F = [&]() { return (const float*)d_in[ii++]; };
  const float* feat8  = F();
  const float* feat16 = F();
  const float* feat32 = F();
  auto CBA  = [&]() { CbaPH c; c.w = F(); c.g = F(); c.b = F(); c.m = F(); c.v = F(); return c; };
  auto CONV = [&]() { ConvPH c; c.w = F(); c.b = F(); return c; };
  auto LIN  = [&]() { LinPH c; c.w = F(); c.b = F(); return c; };
  auto MHA  = [&]() { MhaPH c; c.wi = F(); c.bi = F(); c.wo = F(); c.bo = F(); return c; };
  auto LN   = [&]() { LnPH c; c.g = F(); c.b = F(); return c; };
  CbaPH a08 = CBA(), a16 = CBA(), a32 = CBA();
  CbaPH lat32 = CBA(), lat16 = CBA(), lat08 = CBA();
  CbaPH sm16 = CBA(), sm08 = CBA(), e14a = CBA();
  ConvPH e14b = CONV(), mp1 = CONV(), mp2 = CONV();
  const float* query_embed = F();
  LinPH cls = LIN();
  LayerPH L[6];
  for (int i = 0; i < 6; ++i) {
    L[i].xattn = MHA(); L[i].sattn = MHA();
    L[i].nq = LN(); L[i].nkv = LN(); L[i].n1 = LN(); L[i].n2 = LN();
    L[i].ff1 = LIN(); L[i].ff2 = LIN();
  }
  // img_h, img_w (int scalars, known 512 per reference) — ignored.

  // ---- workspace bump allocator ----
  char* wsb = (char*)d_ws;
  size_t wo = 0;
  auto alloc = [&](size_t bytes) -> void* {
    size_t o = (wo + 255) & ~(size_t)255;
    wo = o + bytes;
    return wsb + o;
  };
  auto tobf = [&](const float* x, long long n) -> __bf16* {
    __bf16* y = (__bf16*)alloc((size_t)n * 2);
    k_cvt_bf16<<<dim3((unsigned)((n + 255) / 256)), dim3(256), 0, stream>>>(x, y, n);
    return y;
  };
  auto bnprep = [&](const CbaPH& c) -> float* {
    float* sb = (float*)alloc(512 * sizeof(float));
    k_bn_prep<<<dim3(1), dim3(256), 0, stream>>>(c.g, c.b, c.m, c.v, sb, sb + 256, 256);
    return sb;
  };
  auto gemm = [&](const GemmP& p, int Z) {
    dim3 g((unsigned)((p.N + 127) / 128), (unsigned)((p.M + 63) / 64), (unsigned)Z);
    k_gemm<<<g, dim3(256), 0, stream>>>(p);
  };

  // ---- bf16 inputs + weights, BN folds ----
  __bf16* f8b   = tobf(feat8,  (long long)B * 1024 * 4096);
  __bf16* f16bb = tobf(feat16, (long long)B * 1024 * 1024);
  __bf16* f32bb = tobf(feat32, (long long)B * 1024 * 256);
  __bf16* w_a08 = tobf(a08.w, 262144);  float* bn_a08 = bnprep(a08);
  __bf16* w_a16 = tobf(a16.w, 262144);  float* bn_a16 = bnprep(a16);
  __bf16* w_a32 = tobf(a32.w, 262144);  float* bn_a32 = bnprep(a32);
  __bf16* w_l32 = tobf(lat32.w, 65536); float* bn_l32 = bnprep(lat32);
  __bf16* w_l16 = tobf(lat16.w, 65536); float* bn_l16 = bnprep(lat16);
  __bf16* w_l08 = tobf(lat08.w, 65536); float* bn_l08 = bnprep(lat08);
  __bf16* w_s16 = tobf(sm16.w, 589824); float* bn_s16 = bnprep(sm16);
  __bf16* w_s08 = tobf(sm08.w, 589824); float* bn_s08 = bnprep(sm08);
  __bf16* w_e4a = tobf(e14a.w, 589824); float* bn_e4a = bnprep(e14a);
  __bf16* w_e4b = tobf(e14b.w, 65536);
  __bf16* w_mp1 = tobf(mp1.w, 589824);
  __bf16* w_mp2 = tobf(mp2.w, 65536);

  // ---- activation buffers ----
  __bf16* p08b = (__bf16*)alloc((size_t)B * 256 * 4096 * 2);
  __bf16* p16b = (__bf16*)alloc((size_t)B * 256 * 1024 * 2);
  __bf16* p32b = (__bf16*)alloc((size_t)B * 256 * 256 * 2);
  float*  l32f = (float*)alloc((size_t)B * 256 * 256 * 4);
  float*  t16f = (float*)alloc((size_t)B * 256 * 1024 * 4);
  float*  l16f = (float*)alloc((size_t)B * 256 * 1024 * 4);
  __bf16* l16b = (__bf16*)alloc((size_t)B * 256 * 1024 * 2);
  float*  l16s = (float*)alloc((size_t)B * 256 * 1024 * 4);
  float*  t08f = (float*)alloc((size_t)B * 256 * 4096 * 4);
  float*  l08f = (float*)alloc((size_t)B * 256 * 4096 * 4);
  __bf16* l08b = (__bf16*)alloc((size_t)B * 256 * 4096 * 2);
  float*  l08s = (float*)alloc((size_t)B * 256 * 4096 * 4);
  __bf16* ebf    = (__bf16*)alloc((size_t)B * 256 * 16384 * 2);
  __bf16* e14ab  = (__bf16*)alloc((size_t)B * 256 * 16384 * 2);
  __bf16* emb14b = (__bf16*)alloc((size_t)B * 256 * 16384 * 2);
  __bf16* prb    = (__bf16*)alloc((size_t)B * 256 * 16384 * 2);
  __bf16* projb  = (__bf16*)alloc((size_t)B * 256 * 16384 * 2);

  auto conv1x1 = [&](const __bf16* W, int K, const __bf16* X, int N,
                     const float* bnS, const float* biasv, int biasMode, int relu,
                     float* Cf, __bf16* Cb) {
    GemmP p{};
    p.A = W; p.lda = K;
    p.B = X; p.bZ = (long long)K * N; p.ldb = N;
    p.C = Cf; p.cZ = (long long)256 * N; p.ldc = N;
    p.Cb = Cb; p.cbZ = (long long)256 * N; p.ldcb = N;
    if (bnS) { p.bnScale = bnS; p.bnShift = bnS + 256; }
    p.bias = biasv; p.biasMode = biasMode; p.relu = relu;
    p.M = 256; p.N = N; p.K = K;
    gemm(p, B);
  };
  auto conv3x3 = [&](const __bf16* W, const __bf16* X, int H, int Wd,
                     const float* bnS, const float* biasv, int biasMode, int relu,
                     float* Cf, __bf16* Cb) {
    int N = H * Wd;
    GemmP p{};
    p.A = W; p.lda = 256 * 9;
    p.B = X; p.bZ = (long long)256 * N; p.ldb = N;
    p.C = Cf; p.cZ = (long long)256 * N; p.ldc = N;
    p.Cb = Cb; p.cbZ = (long long)256 * N; p.ldcb = N;
    if (bnS) { p.bnScale = bnS; p.bnShift = bnS + 256; }
    p.bias = biasv; p.biasMode = biasMode; p.relu = relu;
    p.M = 256; p.N = N; p.K = 256 * 9;
    p.convH = H; p.convW = Wd; p.convKH = 3; p.convKW = 3; p.convPad = 1;
    gemm(p, B);
  };

  // ---- FPN ----
  conv1x1(w_a08, 1024, f8b,  4096, bn_a08, nullptr, 0, 1, nullptr, p08b);
  conv1x1(w_a16, 1024, f16bb, 1024, bn_a16, nullptr, 0, 1, nullptr, p16b);
  conv1x1(w_a32, 1024, f32bb, 256,  bn_a32, nullptr, 0, 1, nullptr, p32b);
  conv1x1(w_l32, 256, p32b, 256,  bn_l32, nullptr, 0, 1, l32f, nullptr);
  conv1x1(w_l16, 256, p16b, 1024, bn_l16, nullptr, 0, 1, t16f, nullptr);
  k_nearest_add<<<dim3((unsigned)(((long long)B * 256 * 1024 + 255) / 256)), dim3(256), 0, stream>>>(
      t16f, l32f, l16f, l16b, B * 256, 32, 32, 16, 16);
  conv3x3(w_s16, l16b, 32, 32, bn_s16, nullptr, 0, 1, l16s, nullptr);
  conv1x1(w_l08, 256, p08b, 4096, bn_l08, nullptr, 0, 1, t08f, nullptr);
  k_nearest_add<<<dim3((unsigned)(((long long)B * 256 * 4096 + 255) / 256)), dim3(256), 0, stream>>>(
      t08f, l16s, l08f, l08b, B * 256, 64, 64, 32, 32);
  conv3x3(w_s08, l08b, 64, 64, bn_s08, nullptr, 0, 1, l08s, nullptr);

  // ---- pixel embedding + mask projection ----
  k_bilinear<<<dim3((unsigned)(((long long)B * 256 * 16384 + 255) / 256)), dim3(256), 0, stream>>>(
      l08s, nullptr, ebf, B * 256, 64, 64, 128, 128);
  conv3x3(w_e4a, ebf, 128, 128, bn_e4a, nullptr, 0, 1, nullptr, e14ab);
  conv1x1(w_e4b, 256, e14ab, 16384, nullptr, e14b.b, 1, 0, nullptr, emb14b);
  conv3x3(w_mp1, emb14b, 128, 128, nullptr, mp1.b, 1, 1, nullptr, prb);
  conv1x1(w_mp2, 256, prb, 16384, nullptr, mp2.b, 1, 0, nullptr, projb);

  // ---- decoder buffers ----
  float*  qf   = (float*)alloc((size_t)B * 100 * 256 * 4);
  __bf16* qb   = (__bf16*)alloc((size_t)B * 100 * 256 * 2);
  __bf16* qnb  = (__bf16*)alloc((size_t)B * 100 * 256 * 2);
  __bf16* qpb  = (__bf16*)alloc((size_t)B * 100 * 256 * 2);
  __bf16* kpb  = (__bf16*)alloc((size_t)B * 4096 * 256 * 2);
  __bf16* vpb  = (__bf16*)alloc((size_t)B * 4096 * 256 * 2);
  __bf16* xob  = (__bf16*)alloc((size_t)B * 100 * 256 * 2);
  float*  scores = (float*)alloc((size_t)16 * 100 * 4096 * 4);
  __bf16* attnb  = (__bf16*)alloc((size_t)16 * 100 * 4096 * 2);
  __bf16* hb     = (__bf16*)alloc((size_t)B * 100 * 1024 * 2);
  __bf16* kvb    = (__bf16*)alloc((size_t)B * 4096 * 256 * 2);
  float*  mlbuf  = (float*)alloc((size_t)B * 100 * 16384 * 4);
  float*  biasbuf = (float*)alloc((size_t)B * 100 * 4096 * 4);

  k_bcast_q<<<dim3((unsigned)((B * 100 * 256 + 255) / 256)), dim3(256), 0, stream>>>(
      query_embed, qf, qb, 100 * 256, B * 100 * 256);

  float* dout = (float*)d_out;
  const long long R_MASKUP = 30200;                     // after class_logits [2,100,151]
  const long long R_ML     = 30200 + 52428800LL;        // after masks_up [2,100,512,512]

  // initial mask logits from initial queries
  {
    GemmP p{};
    p.A = qb; p.aZ = 100 * 256; p.lda = 256;
    p.B = projb; p.bZ = (long long)256 * 16384; p.ldb = 16384;
    p.C = mlbuf; p.cZ = (long long)100 * 16384; p.ldc = 16384;
    p.M = 100; p.N = 16384; p.K = 256;
    gemm(p, B);
  }

  // generic attention (q tokens = 100 per batch)
  auto attention = [&](const MhaPH& mp, const __bf16* wi, const __bf16* wo,
                       const __bf16* kvsrc, long long kvZ, int Nk,
                       const float* biasb, int ldS) {
    GemmP p{};
    // q projection
    p = GemmP{};
    p.A = qnb; p.aZ = 100 * 256; p.lda = 256;
    p.B = wi; p.ldb = 256; p.btrans = 1;
    p.bias = mp.bi; p.biasMode = 2;
    p.Cb = qpb; p.cbZ = 100 * 256; p.ldcb = 256;
    p.M = 100; p.N = 256; p.K = 256;
    gemm(p, B);
    // k projection
    p = GemmP{};
    p.A = kvsrc; p.aZ = kvZ; p.lda = 256;
    p.B = wi + 65536; p.ldb = 256; p.btrans = 1;
    p.bias = mp.bi + 256; p.biasMode = 2;
    p.Cb = kpb; p.cbZ = (long long)Nk * 256; p.ldcb = 256;
    p.M = Nk; p.N = 256; p.K = 256;
    gemm(p, B);
    // v projection
    p = GemmP{};
    p.A = kvsrc; p.aZ = kvZ; p.lda = 256;
    p.B = wi + 131072; p.ldb = 256; p.btrans = 1;
    p.bias = mp.bi + 512; p.biasMode = 2;
    p.Cb = vpb; p.cbZ = (long long)Nk * 256; p.ldcb = 256;
    p.M = Nk; p.N = 256; p.K = 256;
    gemm(p, B);
    // scores per (batch, head)
    for (int b = 0; b < B; ++b) {
      p = GemmP{};
      p.A = qpb + (long long)b * 100 * 256; p.aZ = 32; p.lda = 256;
      p.B = kpb + (long long)b * Nk * 256; p.bZ = 32; p.ldb = 256; p.btrans = 1;
      p.C = scores + (long long)b * 8 * 100 * ldS; p.cZ = (long long)100 * ldS; p.ldc = ldS;
      p.M = 100; p.N = Nk; p.K = 32;
      gemm(p, 8);
    }
    k_softmax<<<dim3(100, 16), dim3(256), 0, stream>>>(
        scores, attnb, (long long)100 * ldS, ldS, Nk, 0.17677669529663687f,
        biasb, (long long)100 * Nk, 8);
    // attn @ V  -> xob (token-major, per-head column slab)
    for (int b = 0; b < B; ++b) {
      p = GemmP{};
      p.A = attnb + (long long)b * 8 * 100 * ldS; p.aZ = (long long)100 * ldS; p.lda = ldS;
      p.B = vpb + (long long)b * Nk * 256; p.bZ = 32; p.ldb = 256;
      p.Cb = xob + (long long)b * 100 * 256; p.cbZ = 32; p.ldcb = 256;
      p.M = 100; p.N = 32; p.K = Nk;
      gemm(p, 8);
    }
    // output projection + residual (updates qf/qb)
    p = GemmP{};
    p.A = xob; p.aZ = 100 * 256; p.lda = 256;
    p.B = wo; p.ldb = 256; p.btrans = 1;
    p.bias = mp.bo; p.biasMode = 2;
    p.residual = qf; p.rZ = 100 * 256; p.ldr = 256;
    p.C = qf; p.cZ = 100 * 256; p.ldc = 256;
    p.Cb = qb; p.cbZ = 100 * 256; p.ldcb = 256;
    p.M = 100; p.N = 256; p.K = 256;
    gemm(p, B);
  };

  struct Lev { const float* f; int h, w; };
  Lev lev[3] = { {l32f, 16, 16}, {l16s, 32, 32}, {l08s, 64, 64} };

  for (int li = 0; li < 6; ++li) {
    const LayerPH& Lp = L[li];
    const Lev f = lev[li % 3];
    const int P = f.h * f.w;

    __bf16* xwi = tobf(Lp.xattn.wi, 196608);
    __bf16* xwo = tobf(Lp.xattn.wo, 65536);
    __bf16* swi = tobf(Lp.sattn.wi, 196608);
    __bf16* swo = tobf(Lp.sattn.wo, 65536);
    __bf16* f1w = tobf(Lp.ff1.w, 262144);
    __bf16* f2w = tobf(Lp.ff2.w, 262144);

    // attention bias from current mask logits at this level
    k_mask_bias<<<dim3(100, B), dim3(256), 0, stream>>>(mlbuf, biasbuf, f.h, f.w);
    // kv layernorm (channel-major -> token-major bf16)
    k_ln_cols<<<dim3(P, B), dim3(256), 0, stream>>>(
        f.f, (long long)256 * P, Lp.nkv.g, Lp.nkv.b, kvb, (long long)P * 256, P);
    // cross-attention
    k_ln_rows<<<dim3(B * 100), dim3(256), 0, stream>>>(qf, Lp.nq.g, Lp.nq.b, qnb);
    attention(Lp.xattn, xwi, xwo, kvb, (long long)P * 256, P, biasbuf, P);
    // self-attention
    k_ln_rows<<<dim3(B * 100), dim3(256), 0, stream>>>(qf, Lp.n1.g, Lp.n1.b, qnb);
    attention(Lp.sattn, swi, swo, qnb, 100 * 256, 100, nullptr, 128);
    // FFN
    k_ln_rows<<<dim3(B * 100), dim3(256), 0, stream>>>(qf, Lp.n2.g, Lp.n2.b, qnb);
    {
      GemmP p{};
      p.A = qnb; p.aZ = 100 * 256; p.lda = 256;
      p.B = f1w; p.ldb = 256; p.btrans = 1;
      p.bias = Lp.ff1.b; p.biasMode = 2; p.relu = 1;
      p.Cb = hb; p.cbZ = (long long)100 * 1024; p.ldcb = 1024;
      p.M = 100; p.N = 1024; p.K = 256;
      gemm(p, B);
    }
    {
      GemmP p{};
      p.A = hb; p.aZ = (long long)100 * 1024; p.lda = 1024;
      p.B = f2w; p.ldb = 1024; p.btrans = 1;
      p.bias = Lp.ff2.b; p.biasMode = 2;
      p.residual = qf; p.rZ = 100 * 256; p.ldr = 256;
      p.C = qf; p.cZ = 100 * 256; p.ldc = 256;
      p.Cb = qb; p.cbZ = 100 * 256; p.ldcb = 256;
      p.M = 100; p.N = 256; p.K = 1024;
      gemm(p, B);
    }
    // refresh mask logits (final iteration writes the output region)
    {
      float* mlo = (li == 5) ? (dout + R_ML) : mlbuf;
      GemmP p{};
      p.A = qb; p.aZ = 100 * 256; p.lda = 256;
      p.B = projb; p.bZ = (long long)256 * 16384; p.ldb = 16384;
      p.C = mlo; p.cZ = (long long)100 * 16384; p.ldc = 16384;
      p.M = 100; p.N = 16384; p.K = 256;
      gemm(p, B);
    }
  }

  // ---- classification head ----
  {
    __bf16* w_cls = tobf(cls.w, 151 * 256);
    GemmP p{};
    p.A = qb; p.aZ = 100 * 256; p.lda = 256;
    p.B = w_cls; p.ldb = 256; p.btrans = 1;
    p.bias = cls.b; p.biasMode = 2;
    p.C = dout; p.cZ = (long long)100 * 151; p.ldc = 151;
    p.M = 100; p.N = 151; p.K = 256;
    gemm(p, B);
  }
  // ---- upsample final mask logits to 512x512 ----
  k_bilinear<<<dim3((unsigned)((200LL * 512 * 512 + 255) / 256)), dim3(256), 0, stream>>>(
      dout + R_ML, dout + R_MASKUP, nullptr, 200, 128, 128, 512, 512);
}
